// RPN_4741643894796
// MI455X (gfx1250) — matrix operand order repaired and checked by reading, not compiled, not run on hardware
//
#include <hip/hip_runtime.h>
#include <hip/hip_bf16.h>
#include <stdint.h>

typedef __attribute__((ext_vector_type(16))) __bf16 v16bf;
typedef __attribute__((ext_vector_type(8)))  float  v8f;

#define H_      64
#define W_      96
#define NPIX    6144        // H_*W_
#define CIN     512
#define A_      9
#define NANCH   55296       // NPIX*A_
#define NSORT   65536
#define PRE_TOPN  6000
#define POST_TOPN 300

// Standard Faster-RCNN base anchors: stride 16, ratios {0.5,1,2} x scales {8,16,32}
__constant__ float BASE_ANCH[9][4] = {
  { -84.f,  -40.f,  99.f,  55.f},
  {-176.f,  -88.f, 191.f, 103.f},
  {-360.f, -184.f, 375.f, 199.f},
  { -56.f,  -56.f,  71.f,  71.f},
  {-120.f, -120.f, 135.f, 135.f},
  {-248.f, -248.f, 263.f, 263.f},
  { -36.f,  -80.f,  51.f,  95.f},
  { -80.f, -168.f,  95.f, 183.f},
  {-168.f, -344.f, 183.f, 359.f}
};

__device__ __forceinline__ unsigned short f2bf(float x) {
  unsigned int u = __float_as_uint(x);
  u += 0x7FFFu + ((u >> 16) & 1u);          // round-to-nearest-even
  return (unsigned short)(u >> 16);
}

// ---------------- prep kernels: f32 -> bf16, re-layout ----------------

__global__ __launch_bounds__(256) void prep_feat(const float* __restrict__ f,
                                                 unsigned short* __restrict__ fhwc) {
  int t = blockIdx.x * 256 + threadIdx.x;
  if (t >= CIN * NPIX) return;
  int ci = t / NPIX, p = t % NPIX;          // src NCHW: f[ci][p]
  fhwc[(long)p * CIN + ci] = f2bf(f[t]);    // dst HWC
}

__global__ __launch_bounds__(256) void prep_wconv(const float* __restrict__ w,
                                                  unsigned short* __restrict__ wt) {
  int t = blockIdx.x * 256 + threadIdx.x;
  if (t >= CIN * CIN * 9) return;
  int co  = t / (CIN * 9);
  int rem = t % (CIN * 9);
  int ci  = rem / 9;
  int pos = rem % 9;
  // B matrix, K-major: row = pos*CIN+ci, col = co
  wt[(long)(pos * CIN + ci) * CIN + co] = f2bf(w[t]);
}

__global__ __launch_bounds__(256) void prep_whead(const float* __restrict__ cls_w,
                                                  const float* __restrict__ bbox_w,
                                                  const float* __restrict__ cls_b,
                                                  const float* __restrict__ bbox_b,
                                                  unsigned short* __restrict__ wh,
                                                  float* __restrict__ hb) {
  int t = blockIdx.x * 256 + threadIdx.x;
  if (t >= CIN * 64) return;
  int ci = t >> 6, n = t & 63;
  float v = 0.f;
  if (n < 18)      v = cls_w [(long)n        * CIN + ci];
  else if (n < 54) v = bbox_w[(long)(n - 18) * CIN + ci];
  wh[(long)ci * 64 + n] = f2bf(v);
  if (t < 64) {
    float b = 0.f;
    if (t < 18)      b = cls_b[t];
    else if (t < 54) b = bbox_b[t - 18];
    hb[t] = b;
  }
}

// ---------------- 3x3 conv as implicit GEMM via WMMA bf16 ----------------
// M = NPIX (6144), N = CIN (512), K = 9*CIN (4608). Workgroup tile 256x64,
// 8 wave32s: each wave computes 32(M) x 64(N) with 8 accumulators ->
// 8 back-to-back v_wmma per K-step from 6 preloaded LDS fragments.

__global__ __launch_bounds__(256) void conv3x3_wmma(const unsigned short* __restrict__ fhwc,
                                                    const unsigned short* __restrict__ wt,
                                                    const float* __restrict__ bias,
                                                    unsigned short* __restrict__ xhwc) {
  __shared__ __align__(16) unsigned short lA[256 * 32];   // [m][k], 16KB
  __shared__ __align__(16) unsigned short lB[32 * 64];    // [k][n], 4KB
  const int tid  = threadIdx.x;
  const int lane = tid & 31;
  const int wave = tid >> 5;
  const int m0 = blockIdx.x * 256;
  const int n0 = blockIdx.y * 64;

  v8f acc[2][4];
  for (int ms = 0; ms < 2; ++ms)
    for (int ns = 0; ns < 4; ++ns)
      for (int e = 0; e < 8; ++e) acc[ms][ns][e] = 0.f;

  // A-tile loader: one thread stages one full 64B row (4 x b128)
  const int apix = m0 + tid;
  const int ay   = apix / W_;
  const int ax   = apix % W_;
  // B-tile loader: 8 threads per row, 8 ushorts (16B) each
  const int brow = tid >> 3;
  const int bseg = tid & 7;
  // fragment lane decomposition
  const int g  = lane >> 4;
  const int mr = lane & 15;

  for (int pos = 0; pos < 9; ++pos) {
    const int sy = ay + (pos / 3) - 1;
    const int sx = ax + (pos % 3) - 1;
    const bool inb = (sy >= 0) && (sy < H_) && (sx >= 0) && (sx < W_);
    const long srow = inb ? ((long)(sy * W_ + sx) * CIN) : 0;
    for (int cc = 0; cc < CIN; cc += 32) {
      __syncthreads();
      // stage A block (256 x 32 bf16)
      uint4 a0 = make_uint4(0, 0, 0, 0), a1 = a0, a2 = a0, a3 = a0;
      if (inb) {
        const unsigned short* src = fhwc + srow + cc;
        a0 = *(const uint4*)(src);
        a1 = *(const uint4*)(src + 8);
        a2 = *(const uint4*)(src + 16);
        a3 = *(const uint4*)(src + 24);
      }
      *(uint4*)&lA[tid * 32 +  0] = a0;
      *(uint4*)&lA[tid * 32 +  8] = a1;
      *(uint4*)&lA[tid * 32 + 16] = a2;
      *(uint4*)&lA[tid * 32 + 24] = a3;
      // stage B block (32 x 64 bf16)
      const unsigned short* bsrc = wt + (long)(pos * CIN + cc + brow) * CIN + n0 + bseg * 8;
      *(uint4*)&lB[brow * 64 + bseg * 8] = *(const uint4*)bsrc;
      __builtin_prefetch(bsrc + 32 * CIN, 0, 0);   // next K-step weight stream
      __syncthreads();

      // preload all fragments, then issue 8 independent WMMAs back-to-back
      union { v16bf v; uint4 q[2]; } af[2], bfr[4];
      for (int ms = 0; ms < 2; ++ms) {
        const int am = wave * 32 + ms * 16 + mr;
        af[ms].q[0] = *(const uint4*)&lA[am * 32 + g * 8];
        af[ms].q[1] = *(const uint4*)&lA[am * 32 + 16 + g * 8];
      }
      const int krow = 8 * g;                       // B: lane -> N col mr, K = 8g+s / 16+8g+(s-8)
      for (int ns = 0; ns < 4; ++ns) {
        for (int s = 0; s < 16; ++s)
          ((unsigned short*)bfr[ns].q)[s] = lB[(krow + (s & 7) + (s >> 3) * 16) * 64 + ns * 16 + mr];
      }
#pragma unroll
      for (int ns = 0; ns < 4; ++ns)
#pragma unroll
        for (int ms = 0; ms < 2; ++ms)
          acc[ms][ns] = __builtin_amdgcn_wmma_f32_16x16x32_bf16(
              false, af[ms].v, false, bfr[ns].v, (short)0, acc[ms][ns], false, false);
    }
  }

  // epilogue: bias + ReLU, store bf16 HWC
  for (int ms = 0; ms < 2; ++ms) {
    for (int ns = 0; ns < 4; ++ns) {
      const int co = n0 + ns * 16 + mr;             // D: N = lane&15
      const float bb = bias[co];
      for (int v = 0; v < 8; ++v) {
        const int pix = m0 + wave * 32 + ms * 16 + v + 8 * g;  // D: M = vgpr + 8*(lane>>4)
        float val = fmaxf(acc[ms][ns][v] + bb, 0.f);
        xhwc[(long)pix * CIN + co] = f2bf(val);
      }
    }
  }
}

// ---------------- 1x1 heads: GEMM M=6144, K=512, N=64 (18 cls + 36 bbox + pad) ----

__global__ __launch_bounds__(256) void heads_wmma(const unsigned short* __restrict__ xhwc,
                                                  const unsigned short* __restrict__ wh,
                                                  const float* __restrict__ hb,
                                                  float* __restrict__ heads) {
  __shared__ __align__(16) unsigned short lA[256 * 32];
  __shared__ __align__(16) unsigned short lB[32 * 64];
  const int tid  = threadIdx.x;
  const int lane = tid & 31;
  const int wave = tid >> 5;
  const int m0 = blockIdx.x * 256;

  v8f acc[2][4];
  for (int ms = 0; ms < 2; ++ms)
    for (int ns = 0; ns < 4; ++ns)
      for (int e = 0; e < 8; ++e) acc[ms][ns][e] = 0.f;

  const int brow = tid >> 3, bseg = tid & 7;
  const int g = lane >> 4, mr = lane & 15;

  for (int cc = 0; cc < CIN; cc += 32) {
    __syncthreads();
    const unsigned short* src = xhwc + (long)(m0 + tid) * CIN + cc;
    *(uint4*)&lA[tid * 32 +  0] = *(const uint4*)(src);
    *(uint4*)&lA[tid * 32 +  8] = *(const uint4*)(src + 8);
    *(uint4*)&lA[tid * 32 + 16] = *(const uint4*)(src + 16);
    *(uint4*)&lA[tid * 32 + 24] = *(const uint4*)(src + 24);
    const unsigned short* bsrc = wh + (long)(cc + brow) * 64 + bseg * 8;
    *(uint4*)&lB[brow * 64 + bseg * 8] = *(const uint4*)bsrc;
    __syncthreads();

    union { v16bf v; uint4 q[2]; } af[2], bfr[4];
    for (int ms = 0; ms < 2; ++ms) {
      const int am = wave * 32 + ms * 16 + mr;
      af[ms].q[0] = *(const uint4*)&lA[am * 32 + g * 8];
      af[ms].q[1] = *(const uint4*)&lA[am * 32 + 16 + g * 8];
    }
    const int krow = 8 * g;
    for (int ns = 0; ns < 4; ++ns) {
      for (int s = 0; s < 16; ++s)
        ((unsigned short*)bfr[ns].q)[s] = lB[(krow + (s & 7) + (s >> 3) * 16) * 64 + ns * 16 + mr];
    }
#pragma unroll
    for (int ns = 0; ns < 4; ++ns)
#pragma unroll
      for (int ms = 0; ms < 2; ++ms)
        acc[ms][ns] = __builtin_amdgcn_wmma_f32_16x16x32_bf16(
            false, af[ms].v, false, bfr[ns].v, (short)0, acc[ms][ns], false, false);
  }

  for (int ms = 0; ms < 2; ++ms) {
    for (int ns = 0; ns < 4; ++ns) {
      const int n  = ns * 16 + mr;
      const float bb = hb[n];
      for (int v = 0; v < 8; ++v) {
        const int pix = m0 + wave * 32 + ms * 16 + v + 8 * g;
        heads[(long)pix * 64 + n] = acc[ms][ns][v] + bb;
      }
    }
  }
}

// ---------------- proposals: softmax score, box decode, clip, filter, sort keys ----

__global__ __launch_bounds__(256) void proposal_kernel(const float* __restrict__ heads,
                                                       const float* __restrict__ iminfo,
                                                       float* __restrict__ boxes,
                                                       unsigned long long* __restrict__ keys) {
  int t = blockIdx.x * 256 + threadIdx.x;
  if (t >= NSORT) return;
  if (t >= NANCH) { keys[t] = 0ull; return; }
  const int p = t / A_, a = t % A_;

  const float sbg = heads[(long)p * 64 + a];
  const float sfg = heads[(long)p * 64 + 9 + a];
  const float score = 1.f / (1.f + expf(sbg - sfg));   // softmax fg prob

  const float dx = heads[(long)p * 64 + 18 + a * 4 + 0];
  const float dy = heads[(long)p * 64 + 18 + a * 4 + 1];
  const float dw = heads[(long)p * 64 + 18 + a * 4 + 2];
  const float dh = heads[(long)p * 64 + 18 + a * 4 + 3];

  const float shx = (float)((p % W_) * 16);
  const float shy = (float)((p / W_) * 16);
  const float ax1 = BASE_ANCH[a][0] + shx, ay1 = BASE_ANCH[a][1] + shy;
  const float ax2 = BASE_ANCH[a][2] + shx, ay2 = BASE_ANCH[a][3] + shy;
  const float aw  = ax2 - ax1 + 1.f, ahh = ay2 - ay1 + 1.f;
  const float acx = ax1 + 0.5f * aw, acy = ay1 + 0.5f * ahh;

  const float cx = dx * aw + acx, cy = dy * ahh + acy;
  const float pw = expf(dw) * aw, ph = expf(dh) * ahh;

  const float imh = iminfo[0], imw = iminfo[1], scal = iminfo[2];
  float x1 = fminf(fmaxf(cx - 0.5f * pw, 0.f), imw - 1.f);
  float y1 = fminf(fmaxf(cy - 0.5f * ph, 0.f), imh - 1.f);
  float x2 = fminf(fmaxf(cx + 0.5f * pw, 0.f), imw - 1.f);
  float y2 = fminf(fmaxf(cy + 0.5f * ph, 0.f), imh - 1.f);

  boxes[t * 4 + 0] = x1; boxes[t * 4 + 1] = y1;
  boxes[t * 4 + 2] = x2; boxes[t * 4 + 3] = y2;

  const float ms = 16.f * scal;
  const bool keep = (x2 - x1 + 1.f >= ms) && (y2 - y1 + 1.f >= ms);
  const float s = keep ? score : __uint_as_float(0xFF800000u);  // -inf
  const unsigned int b = __float_as_uint(s);
  const unsigned int ord = (b & 0x80000000u) ? ~b : (b | 0x80000000u);
  keys[t] = ((unsigned long long)ord << 32) | (unsigned int)(~(unsigned int)t);
}

// ---------------- bitonic sort (descending), one stage per launch ----------------

__global__ __launch_bounds__(256) void bitonic_step(unsigned long long* __restrict__ keys,
                                                    int k, int j) {
  int i = blockIdx.x * 256 + threadIdx.x;
  int l = i ^ j;
  if (l > i) {
    unsigned long long a = keys[i], b = keys[l];
    bool up = ((i & k) == 0);                 // descending in 'up' blocks
    if (up ? (a < b) : (a > b)) { keys[i] = b; keys[l] = a; }
  }
}

// ---------------- sequential NMS: one workgroup, everything in 320KB LDS ----------

__global__ __launch_bounds__(1024) void nms_kernel(const unsigned long long* __restrict__ keys,
                                                   const float* __restrict__ boxes,
                                                   float* __restrict__ out) {
  __shared__ float sbx[PRE_TOPN * 4];        // 96 KB
  __shared__ float sarea[PRE_TOPN];          // 24 KB
  __shared__ unsigned char ssup[PRE_TOPN];
  __shared__ unsigned char sval[PRE_TOPN];
  __shared__ int sred[1024];
  __shared__ float pick[5];
  __shared__ int spick;

  const int j = threadIdx.x;
  for (int si = j; si < PRE_TOPN; si += 1024) {
    unsigned long long k = keys[si];
    unsigned int ord = (unsigned int)(k >> 32);
    unsigned int idx = ~(unsigned int)k;
    float x1 = 0.f, y1 = 0.f, x2 = 0.f, y2 = 0.f;
    if (idx < NANCH) {
      x1 = boxes[idx * 4 + 0]; y1 = boxes[idx * 4 + 1];
      x2 = boxes[idx * 4 + 2]; y2 = boxes[idx * 4 + 3];
    }
    sbx[si * 4 + 0] = x1; sbx[si * 4 + 1] = y1;
    sbx[si * 4 + 2] = x2; sbx[si * 4 + 3] = y2;
    sarea[si] = (x2 - x1 + 1.f) * (y2 - y1 + 1.f);
    ssup[si] = 0;
    sval[si] = (ord > 0x007FFFFFu) ? 1 : 0;   // finite score (> -inf)
  }
  __syncthreads();

  for (int it = 0; it < POST_TOPN; ++it) {
    // argmax over masked scores == first unsuppressed valid entry in sorted order
    int local = 0x7FFFFFFF;
    for (int si = j; si < PRE_TOPN; si += 1024)
      if (!ssup[si] && sval[si]) { local = si; break; }
    sred[j] = local;
    __syncthreads();
    for (int s = 512; s > 0; s >>= 1) {
      if (j < s) { int o = sred[j + s]; if (o < sred[j]) sred[j] = o; }
      __syncthreads();
    }
    if (j == 0) {
      int i = sred[0];
      const bool valid = (i != 0x7FFFFFFF);
      if (!valid) i = 0;                     // argmax of all -inf -> index 0
      spick = i;
      pick[0] = sbx[i * 4 + 0]; pick[1] = sbx[i * 4 + 1];
      pick[2] = sbx[i * 4 + 2]; pick[3] = sbx[i * 4 + 3];
      pick[4] = valid ? 1.f : 0.f;
    }
    __syncthreads();
    const float bx1 = pick[0], by1 = pick[1], bx2 = pick[2], by2 = pick[3];
    const float barea = sarea[spick];
    for (int si = j; si < PRE_TOPN; si += 1024) {
      float xx1 = fmaxf(bx1, sbx[si * 4 + 0]);
      float yy1 = fmaxf(by1, sbx[si * 4 + 1]);
      float xx2 = fminf(bx2, sbx[si * 4 + 2]);
      float yy2 = fminf(by2, sbx[si * 4 + 3]);
      float inter = fmaxf(xx2 - xx1 + 1.f, 0.f) * fmaxf(yy2 - yy1 + 1.f, 0.f);
      float iou = inter / (barea + sarea[si] - inter);
      if (iou > 0.7f) ssup[si] = 1;
    }
    if (j == 0) {
      const bool v = pick[4] > 0.f;
      out[it * 5 + 0] = 0.f;
      out[it * 5 + 1] = v ? pick[0] : 0.f;
      out[it * 5 + 2] = v ? pick[1] : 0.f;
      out[it * 5 + 3] = v ? pick[2] : 0.f;
      out[it * 5 + 4] = v ? pick[3] : 0.f;
    }
    __syncthreads();
  }
}

// ---------------- launcher ----------------

extern "C" void kernel_launch(void* const* d_in, const int* in_sizes, int n_in,
                              void* d_out, int out_size, void* d_ws, size_t ws_size,
                              hipStream_t stream) {
  const float* feat   = (const float*)d_in[0];
  const float* iminfo = (const float*)d_in[1];
  const float* conv_w = (const float*)d_in[2];
  const float* conv_b = (const float*)d_in[3];
  const float* cls_w  = (const float*)d_in[4];
  const float* cls_b  = (const float*)d_in[5];
  const float* bbox_w = (const float*)d_in[6];
  const float* bbox_b = (const float*)d_in[7];

  char* base = (char*)d_ws;
  size_t off = 0;
  auto carve = [&](size_t bytes) -> void* {
    void* p = base + off;
    off = (off + bytes + 255) & ~(size_t)255;
    return p;
  };

  unsigned short* fhwc = (unsigned short*)carve((size_t)NPIX * CIN * 2);
  unsigned short* wt   = (unsigned short*)carve((size_t)9 * CIN * CIN * 2);
  unsigned short* xh   = (unsigned short*)carve((size_t)NPIX * CIN * 2);
  unsigned short* wh   = (unsigned short*)carve((size_t)CIN * 64 * 2);
  float*          hb   = (float*)carve(64 * 4);
  float*          hd   = (float*)carve((size_t)NPIX * 64 * 4);
  float*          bxs  = (float*)carve((size_t)NANCH * 4 * 4);
  unsigned long long* keys = (unsigned long long*)carve((size_t)NSORT * 8);

  prep_feat <<<(CIN * NPIX + 255) / 256, 256, 0, stream>>>(feat, fhwc);
  prep_wconv<<<(CIN * CIN * 9 + 255) / 256, 256, 0, stream>>>(conv_w, wt);
  prep_whead<<<(CIN * 64 + 255) / 256, 256, 0, stream>>>(cls_w, bbox_w, cls_b, bbox_b, wh, hb);

  conv3x3_wmma<<<dim3(NPIX / 256, CIN / 64), 256, 0, stream>>>(fhwc, wt, conv_b, xh);
  heads_wmma  <<<NPIX / 256, 256, 0, stream>>>(xh, wh, hb, hd);

  proposal_kernel<<<NSORT / 256, 256, 0, stream>>>(hd, iminfo, bxs, keys);

  for (int k = 2; k <= NSORT; k <<= 1)
    for (int jj = k >> 1; jj >= 1; jj >>= 1)
      bitonic_step<<<NSORT / 256, 256, 0, stream>>>(keys, k, jj);

  nms_kernel<<<1, 1024, 0, stream>>>(keys, bxs, (float*)d_out);
}
